// MultiAttModel_17532056502539
// MI455X (gfx1250) — matrix operand order repaired and checked by reading, not compiled, not run on hardware
//
#include <hip/hip_runtime.h>
#include <hip/hip_bf16.h>

// ---------------- model constants ----------------
#define BB      64
#define DIM     128
#define LL      2048
#define HH      8
#define WIN     5
#define KW      11
#define DD      2048          // 2*H*DIM
#define CO_TOT  1152          // DIM + H*DIM
#define KKT     1408          // DIM*KW  (= 44 * 32)
#define NKSTEP  44
#define LP      2064          // padded L stride (L + 2*WIN rounded to 16)
#define LCHUNK  256
#define XMAX    272           // staged columns per chunk (256 + 10 halo, padded)
#define CISTR   136           // transposed tile ci-stride (halfwords): 272B rows, bank-tiled
#define CSTR    264           // C tile col stride (floats)
#define EPSF    1e-5f
#define SLOPEF  0.01f

typedef float v8f   __attribute__((ext_vector_type(8)));
typedef __bf16 v16bf __attribute__((ext_vector_type(16)));

union BF16Frag { unsigned short s[16]; uint4 u4[2]; v16bf v; };
union U4H8    { uint4 u; unsigned short s[8]; };

// native f32 -> bf16 (RNE) : lowers to v_cvt_bf16_f32 / v_cvt_pk_bf16_f32
__device__ __forceinline__ unsigned short f2bf(float f) {
    __bf16 h = (__bf16)f;
    return __builtin_bit_cast(unsigned short, h);
}
__device__ __forceinline__ float bf2f(unsigned short s) {
    return __uint_as_float(((unsigned)s) << 16);
}

// ---------------- packing kernels ----------------
__global__ void pack_emb_kernel(const float* __restrict__ emb, unsigned short* __restrict__ embp) {
    const long total = (long)BB * DIM * LP;
    for (long idx = (long)blockIdx.x * blockDim.x + threadIdx.x; idx < total;
         idx += (long)gridDim.x * blockDim.x) {
        int  lp = (int)(idx % LP);
        long bc = idx / LP;
        int  l  = lp - WIN;
        float x = (l >= 0 && l < LL) ? emb[bc * LL + l] : 0.0f;
        embp[idx] = f2bf(x);
    }
}

// kk = kw*128 + ci ; co 0..127 = cnn_v_w, co 128..1151 = wq_w (flattened H*DIM)
__global__ void pack_w_kernel(const float* __restrict__ cnn_v_w, const float* __restrict__ wq_w,
                              unsigned short* __restrict__ wpack) {
    const long total = (long)CO_TOT * KKT;
    for (long idx = (long)blockIdx.x * blockDim.x + threadIdx.x; idx < total;
         idx += (long)gridDim.x * blockDim.x) {
        int kk = (int)(idx % KKT);
        int co = (int)(idx / KKT);
        int kw = kk >> 7;
        int ci = kk & 127;
        float x = (co < DIM) ? cnn_v_w[((long)co * DIM + ci) * KW + kw]
                             : wq_w[((long)(co - DIM) * DIM + ci) * KW + kw];
        wpack[idx] = f2bf(x);
    }
}

__global__ void pack_rdw_kernel(const float* __restrict__ rd_w, unsigned short* __restrict__ rdwb) {
    const long total = 3L * DD * DD;
    for (long idx = (long)blockIdx.x * blockDim.x + threadIdx.x; idx < total;
         idx += (long)gridDim.x * blockDim.x)
        rdwb[idx] = f2bf(rd_w[idx]);
}

// -------- shared helper: stage transposed input slab RAWT[x][ci] = embp[b][ci][l0+x] --------
__device__ __forceinline__ void stage_rawt(unsigned short* RAWT, const unsigned short* erow,
                                           int l0, int tid) {
    // 128 ci rows x (XMAX/8) uint4 segments; lane ci = tid -> LDS stores are
    // contiguous halfwords across lanes (conflict-free).
    const int ci = tid & 127;
    for (int q = 0; q < XMAX / 8; ++q) {
        U4H8 d;
        d.u = ((const uint4*)(erow + (long)ci * LP + l0))[q];
#pragma unroll
        for (int j = 0; j < 8; ++j)
            RAWT[(q * 8 + j) * CISTR + ci] = d.s[j];
    }
}

// ---------------- values conv (WMMA bf16) ----------------
// grid = B * (DIM/16), block = 128 (4 waves). Each wg: 16 out-channels, full L.
__global__ __launch_bounds__(128)
void conv_values_kernel(const unsigned short* __restrict__ wpack,
                        const unsigned short* __restrict__ embp,
                        const float* __restrict__ cnn_v_b,
                        unsigned short* __restrict__ values) {
    extern __shared__ char smem[];
    unsigned short* Alds = (unsigned short*)smem;                  // [16][KKT] : 45056 B
    unsigned short* RAWT = (unsigned short*)(smem + 45056);        // [272][136]: 73984 B

    const int b      = blockIdx.x >> 3;
    const int co0    = (blockIdx.x & 7) * 16;
    const int tid    = threadIdx.x;
    const int lane   = tid & 31;
    const int wave   = tid >> 5;
    const int laneN  = lane & 15;
    const int hi     = lane >> 4;
    const int abase  = hi * 8;    // C-row base (acc element r -> row r + abase)
    const int bbase  = hi * 16;   // B-frag K base

    { // preload 16 weight rows (contiguous in wpack)
        const uint4* src = (const uint4*)(wpack + (long)co0 * KKT);
        uint4* dst = (uint4*)Alds;
        for (int i = tid; i < (16 * KKT) / 8; i += 128) dst[i] = src[i];
    }
    const unsigned short* erow = embp + (long)b * DIM * LP;

    for (int chunk = 0; chunk < LL / LCHUNK; ++chunk) {
        const int l0 = chunk * LCHUNK;
        __syncthreads();
        stage_rawt(RAWT, erow, l0, tid);
        __syncthreads();

        v8f acc[4];
#pragma unroll
        for (int t = 0; t < 4; ++t)
#pragma unroll
            for (int r = 0; r < 8; ++r) acc[t][r] = 0.0f;

        for (int ks = 0; ks < NKSTEP; ++ks) {
            const int kw  = ks >> 2;
            const int ci0 = (ks & 3) << 5;
            const int kk0 = ks << 5;
            BF16Frag af;
            const uint4* ap = (const uint4*)(Alds + laneN * KKT + kk0 + bbase);
            af.u4[0] = ap[0];
            af.u4[1] = ap[1]; // +8 halfwords
#pragma unroll
            for (int t = 0; t < 4; ++t) {
                const int x = wave * 64 + t * 16 + laneN + kw;
                const uint4* bp = (const uint4*)(RAWT + x * CISTR + ci0 + bbase);
                BF16Frag bf;
                bf.u4[0] = bp[0];
                bf.u4[1] = bp[1];   // 16 contiguous halfwords (transposed slab)
                acc[t] = __builtin_amdgcn_wmma_f32_16x16x32_bf16(
                    false, af.v, false, bf.v, (short)0, acc[t], false, false);
            }
        }
#pragma unroll
        for (int t = 0; t < 4; ++t) {
            const int n = wave * 64 + t * 16 + laneN;
#pragma unroll
            for (int r = 0; r < 8; ++r) {
                const int m = r + abase;
                float v = acc[t][r] + cnn_v_b[co0 + m];
                values[((long)(b * DIM + co0 + m)) * LL + l0 + n] = f2bf(v);
            }
        }
    }
}

// ---------------- heads conv + online softmax + v*w reduce + BN0 ----------------
// grid = B * (H*DIM/16) = 64*64, block = 128.
__global__ __launch_bounds__(128)
void conv_heads_kernel(const unsigned short* __restrict__ wpack,
                       const unsigned short* __restrict__ embp,
                       const unsigned short* __restrict__ values,
                       const float* __restrict__ bn0_g, const float* __restrict__ bn0_b,
                       const float* __restrict__ bn0_m, const float* __restrict__ bn0_v,
                       float* __restrict__ cat0) {
    extern __shared__ char smem[];
    unsigned short* Alds = (unsigned short*)smem;                     // 45056 B
    unsigned short* RAWT = (unsigned short*)(smem + 45056);           // 73984 B
    float* Ctile  = (float*)(smem + 119040);                          // 16*264*4 = 16896 B
    float* redmax = (float*)(smem + 135936);                          // 128 f
    float* redS   = (float*)(smem + 136448);
    float* redA   = (float*)(smem + 136960);
    float* redM   = (float*)(smem + 137472);
    float* stM    = (float*)(smem + 137984);                          // 16 f each
    float* stS    = (float*)(smem + 138048);
    float* stA    = (float*)(smem + 138112);
    float* stMX   = (float*)(smem + 138176);

    const int b    = blockIdx.x >> 6;
    const int tile = blockIdx.x & 63;
    const int co0  = DIM + tile * 16;       // global conv out-channel base
    const int h    = tile >> 3;
    const int dim0 = (tile & 7) * 16;

    const int tid   = threadIdx.x;
    const int lane  = tid & 31;
    const int wave  = tid >> 5;
    const int laneN = lane & 15;
    const int hi    = lane >> 4;
    const int abase = hi * 8;
    const int bbase = hi * 16;

    {
        const uint4* src = (const uint4*)(wpack + (long)co0 * KKT);
        uint4* dst = (uint4*)Alds;
        for (int i = tid; i < (16 * KKT) / 8; i += 128) dst[i] = src[i];
    }
    if (tid < 16) { stM[tid] = -3.4e38f; stS[tid] = 0.f; stA[tid] = 0.f; stMX[tid] = 0.f; }
    const unsigned short* erow = embp + (long)b * DIM * LP;
    const int rrow = tid >> 3;   // reduction row 0..15
    const int seg  = tid & 7;    // 8 segments of 32 cols

    for (int chunk = 0; chunk < LL / LCHUNK; ++chunk) {
        const int l0 = chunk * LCHUNK;
        __syncthreads();
        stage_rawt(RAWT, erow, l0, tid);
        __syncthreads();

        v8f acc[4];
#pragma unroll
        for (int t = 0; t < 4; ++t)
#pragma unroll
            for (int r = 0; r < 8; ++r) acc[t][r] = 0.0f;

        for (int ks = 0; ks < NKSTEP; ++ks) {
            const int kw  = ks >> 2;
            const int ci0 = (ks & 3) << 5;
            const int kk0 = ks << 5;
            BF16Frag af;
            const uint4* ap = (const uint4*)(Alds + laneN * KKT + kk0 + bbase);
            af.u4[0] = ap[0];
            af.u4[1] = ap[1];
#pragma unroll
            for (int t = 0; t < 4; ++t) {
                const int x = wave * 64 + t * 16 + laneN + kw;
                const uint4* bp = (const uint4*)(RAWT + x * CISTR + ci0 + bbase);
                BF16Frag bf;
                bf.u4[0] = bp[0];
                bf.u4[1] = bp[1];
                acc[t] = __builtin_amdgcn_wmma_f32_16x16x32_bf16(
                    false, af.v, false, bf.v, (short)0, acc[t], false, false);
            }
        }
        // spill chunk to LDS (softmax is per-row: bias is constant per row -> dropped)
#pragma unroll
        for (int t = 0; t < 4; ++t) {
            const int n = wave * 64 + t * 16 + laneN;
#pragma unroll
            for (int r = 0; r < 8; ++r)
                Ctile[(r + abase) * CSTR + n] = acc[t][r];
        }
        __syncthreads();

        // pass 1: chunk row max
        float lm = -3.4e38f;
        for (int c = seg * 32; c < seg * 32 + 32; ++c)
            lm = fmaxf(lm, Ctile[rrow * CSTR + c]);
        redmax[tid] = lm;
        __syncthreads();
        float cm = -3.4e38f;
#pragma unroll
        for (int j = 0; j < 8; ++j) cm = fmaxf(cm, redmax[rrow * 8 + j]);

        // pass 2: exp-sums with values
        float ps = 0.f, pa = 0.f, pmx = -3.4e38f;
        const unsigned short* vrow = values + ((long)(b * DIM + dim0 + rrow)) * LL + l0;
        for (int c = seg * 32; c < seg * 32 + 32; ++c) {
            float e = __expf(Ctile[rrow * CSTR + c] - cm);
            float p = bf2f(vrow[c]) * e;
            ps += e; pa += p; pmx = fmaxf(pmx, p);
        }
        redS[tid] = ps; redA[tid] = pa; redM[tid] = pmx;
        __syncthreads();
        if (seg == 0) {
            float cs = 0.f, ca = 0.f, cmx = -3.4e38f;
#pragma unroll
            for (int j = 0; j < 8; ++j) {
                cs += redS[rrow * 8 + j];
                ca += redA[rrow * 8 + j];
                cmx = fmaxf(cmx, redM[rrow * 8 + j]);
            }
            if (chunk == 0) {
                stM[rrow] = cm; stS[rrow] = cs; stA[rrow] = ca; stMX[rrow] = cmx;
            } else {
                float m0 = stM[rrow];
                float nm = fmaxf(m0, cm);
                float f1 = __expf(m0 - nm), f2 = __expf(cm - nm);
                stS[rrow]  = stS[rrow] * f1 + cs * f2;
                stA[rrow]  = stA[rrow] * f1 + ca * f2;
                stMX[rrow] = fmaxf(stMX[rrow] * f1, cmx * f2);
                stM[rrow]  = nm;
            }
        }
    }
    __syncthreads();
    if (tid < 16) {
        const int r   = tid;
        const float s = stS[r];
        const float xs = stA[r] / s;
        const float xm = stMX[r] / s;
        const int i1 = h * DIM + dim0 + r;          // x_sum slot
        const int i2 = 1024 + i1;                   // x_max slot
        cat0[b * DD + i1] = (xs - bn0_m[i1]) * bn0_g[i1] * rsqrtf(bn0_v[i1] + EPSF) + bn0_b[i1];
        cat0[b * DD + i2] = (xm - bn0_m[i2]) * bn0_g[i2] * rsqrtf(bn0_v[i2] + EPSF) + bn0_b[i2];
    }
}

// ---------------- residual dense layer: out = in + leaky(bn(in @ W^T + b)) ----------------
// grid = DD/64 = 32 blocks, block = 128 (4 waves, each wave = one 16-row M strip).
__global__ __launch_bounds__(128)
void rd_layer_kernel(const float* __restrict__ cat_in,
                     const unsigned short* __restrict__ wb,   // [DD][DD] bf16 (row n, contiguous k)
                     const float* __restrict__ bias,
                     const float* __restrict__ g, const float* __restrict__ bb,
                     const float* __restrict__ mm, const float* __restrict__ vv,
                     float* __restrict__ cat_out) {
    extern __shared__ char smem[];
    unsigned short* Alds = (unsigned short*)smem;   // [64][136] bf16
    const int ASTR = 136;

    const int Ncol0 = blockIdx.x * 64;
    const int tid   = threadIdx.x;
    const int lane  = tid & 31;
    const int wave  = tid >> 5;
    const int laneN = lane & 15;
    const int hi    = lane >> 4;
    const int abase = hi * 8;
    const int bbk   = hi * 16;
    const int i0    = wave * 16;

    v8f acc[4];
#pragma unroll
    for (int t = 0; t < 4; ++t)
#pragma unroll
        for (int r = 0; r < 8; ++r) acc[t][r] = 0.0f;

    for (int slab = 0; slab < DD / 128; ++slab) {
        __syncthreads();
        { // stage+convert A slab: 64 rows x 128 k
            int row = tid >> 1, half = tid & 1;
            const float* src = cat_in + (long)row * DD + slab * 128 + half * 64;
            unsigned short* dst = Alds + row * ASTR + half * 64;
            for (int c = 0; c < 64; ++c) dst[c] = f2bf(src[c]);
        }
        __syncthreads();
#pragma unroll
        for (int ksub = 0; ksub < 4; ++ksub) {
            const int k0    = ksub * 32;
            const int kglob = slab * 128 + k0;
            BF16Frag af;
            const uint4* ap = (const uint4*)(Alds + (i0 + laneN) * ASTR + k0 + bbk);
            af.u4[0] = ap[0];
            af.u4[1] = ap[1];
#pragma unroll
            for (int t = 0; t < 4; ++t) {
                const int nglob = Ncol0 + t * 16 + laneN;
                const uint4* bp = (const uint4*)(wb + (long)nglob * DD + kglob + bbk);
                BF16Frag bf;
                bf.u4[0] = bp[0];
                bf.u4[1] = bp[1];   // 16 contiguous halfwords
                acc[t] = __builtin_amdgcn_wmma_f32_16x16x32_bf16(
                    false, af.v, false, bf.v, (short)0, acc[t], false, false);
            }
        }
    }
#pragma unroll
    for (int t = 0; t < 4; ++t) {
        const int nglob = Ncol0 + t * 16 + laneN;
        const float bi = bias[nglob];
        const float sc = g[nglob] * rsqrtf(vv[nglob] + EPSF);
        const float sb = bb[nglob];
        const float mu = mm[nglob];
#pragma unroll
        for (int r = 0; r < 8; ++r) {
            const int i = i0 + r + abase;
            float x = acc[t][r] + bi;
            float y = (x - mu) * sc + sb;
            y = (y >= 0.f) ? y : SLOPEF * y;
            cat_out[(long)i * DD + nglob] = cat_in[(long)i * DD + nglob] + y;
        }
    }
}

// ---------------- final projection ----------------
__global__ void out_layer_kernel(const float* __restrict__ cat, const float* __restrict__ out_w,
                                 const float* __restrict__ out_b, float* __restrict__ out) {
    __shared__ float red[128];
    const int i = blockIdx.x, t = threadIdx.x;
    float p = 0.f;
    for (int n = t; n < DD; n += 128) p += cat[(long)i * DD + n] * out_w[n];
    red[t] = p;
    __syncthreads();
    for (int s = 64; s > 0; s >>= 1) {
        if (t < s) red[t] += red[t + s];
        __syncthreads();
    }
    if (t == 0) out[i] = red[0] + out_b[0];
}

// ---------------- launcher ----------------
extern "C" void kernel_launch(void* const* d_in, const int* in_sizes, int n_in,
                              void* d_out, int out_size, void* d_ws, size_t ws_size,
                              hipStream_t stream) {
    const float* emb      = (const float*)d_in[0];
    const float* cnn_v_w  = (const float*)d_in[1];
    const float* cnn_v_b  = (const float*)d_in[2];
    const float* wq_w     = (const float*)d_in[3];
    // d_in[4] = wq_b : softmax over L is invariant to a per-row constant -> unused
    const float* bn0_g    = (const float*)d_in[5];
    const float* bn0_b    = (const float*)d_in[6];
    const float* bn0_m    = (const float*)d_in[7];
    const float* bn0_v    = (const float*)d_in[8];
    const float* rd_w     = (const float*)d_in[9];
    const float* rd_b     = (const float*)d_in[10];
    const float* rd_bn_g  = (const float*)d_in[11];
    const float* rd_bn_b  = (const float*)d_in[12];
    const float* rd_bn_m  = (const float*)d_in[13];
    const float* rd_bn_v  = (const float*)d_in[14];
    const float* out_w    = (const float*)d_in[15];
    const float* out_b    = (const float*)d_in[16];
    float* out            = (float*)d_out;

    char* ws = (char*)d_ws;
    unsigned short* embp  = (unsigned short*)(ws);                 // 64*128*2064*2   = 33,816,576
    unsigned short* wpack = (unsigned short*)(ws + 33816576);      // 1152*1408*2     =  3,244,032
    unsigned short* rdwb  = (unsigned short*)(ws + 37060608);      // 3*2048*2048*2   = 25,165,824
    unsigned short* vals  = (unsigned short*)(ws + 62226432);      // 64*128*2048*2   = 33,554,432
    float*          cat0  = (float*)(ws + 95780864);               // 64*2048*4       =    524,288
    float*          cat1  = (float*)(ws + 96305152);               // 64*2048*4

    // 1) pack inputs/weights to bf16
    pack_emb_kernel<<<2048, 256, 0, stream>>>(emb, embp);
    pack_w_kernel<<<1024, 256, 0, stream>>>(cnn_v_w, wq_w, wpack);
    pack_rdw_kernel<<<2048, 256, 0, stream>>>(rd_w, rdwb);

    // 2) values conv (bf16 WMMA, f32 accumulate)
    conv_values_kernel<<<BB * (DIM / 16), 128, 119040, stream>>>(wpack, embp, cnn_v_b, vals);

    // 3) heads conv + fused online-softmax + sum/max reduce + BN0
    conv_heads_kernel<<<BB * (HH * DIM / 16), 128, 138240, stream>>>(
        wpack, embp, vals, bn0_g, bn0_b, bn0_m, bn0_v, cat0);

    // 4) three residual dense layers (ping-pong cat buffers)
    rd_layer_kernel<<<DD / 64, 128, 64 * 136 * 2, stream>>>(
        cat0, rdwb + 0L * DD * DD, rd_b + 0 * DD,
        rd_bn_g + 0 * DD, rd_bn_b + 0 * DD, rd_bn_m + 0 * DD, rd_bn_v + 0 * DD, cat1);
    rd_layer_kernel<<<DD / 64, 128, 64 * 136 * 2, stream>>>(
        cat1, rdwb + 1L * DD * DD, rd_b + 1 * DD,
        rd_bn_g + 1 * DD, rd_bn_b + 1 * DD, rd_bn_m + 1 * DD, rd_bn_v + 1 * DD, cat0);
    rd_layer_kernel<<<DD / 64, 128, 64 * 136 * 2, stream>>>(
        cat0, rdwb + 2L * DD * DD, rd_b + 2 * DD,
        rd_bn_g + 2 * DD, rd_bn_b + 2 * DD, rd_bn_m + 2 * DD, rd_bn_v + 2 * DD, cat1);

    // 5) final projection -> [B, 1]
    out_layer_kernel<<<BB, 128, 0, stream>>>(cat1, out_w, out_b, out);
}
